// SwinTransformerBlock_68521908240549
// MI455X (gfx1250) — compile-verified, matrix operand order checked by reading, not run on hardware
//
#include <hip/hip_runtime.h>

typedef __attribute__((ext_vector_type(16))) _Float16 v16h;
typedef __attribute__((ext_vector_type(8)))  float    v8f;

#define DIMC   192
#define HEADS  6
#define HD     32
#define WS_    7
#define SS_    3
#define DD     28
#define NT     343          // tokens per window
#define NP     352          // padded to 22*16
#define NWIN   64           // windows per batch
#define BW     128          // total windows (B=2)
#define LTOK   21952
#define BB     2
#define HID    768
#define SCALE_F 0.17677669529663687f   // 32^-0.5
#define NEG_INF (-1e30f)

// ---- WMMA tile helper, NB-way N-blocked + double-buffered operand fetch:
// cs[j](16x16,f32) = A[16xK] * Bt[(nbase+16j)..+16, K]^T    for j in [0,NB)
// A row-major [M x K], Bt row-major [N x K]. Per-lane: lane&15 = row(A)/col(B),
// lane>>4 = K-half; one contiguous 32B v16h per operand per k-chunk.
// Operands for chunk t+1 are prefetched into fresh registers while the NB
// WMMAs of chunk t execute, so VMEM latency overlaps the matrix pipe.
template <int K, int NB>
__device__ __forceinline__ void wmma_tile_nb(const _Float16* __restrict__ A, int lda,
                                             const _Float16* __restrict__ Bt, int ldb,
                                             int mbase, int nbase, int lane, v8f* cs) {
  const _Float16* ap = A  + (size_t)(mbase + (lane & 15)) * lda + ((lane >> 4) << 4);
  const _Float16* bp = Bt + (size_t)(nbase + (lane & 15)) * ldb + ((lane >> 4) << 4);
  v8f z = {};
#pragma unroll
  for (int j = 0; j < NB; j++) cs[j] = z;
  v16h a = *(const v16h*)(ap);
  v16h b[NB];
#pragma unroll
  for (int j = 0; j < NB; j++) b[j] = *(const v16h*)(bp + (size_t)(j * 16) * ldb);
#pragma unroll
  for (int kb = 0; kb < K; kb += 32) {
    v16h an;
    v16h bn[NB];
    if (kb + 32 < K) {                       // prefetch next chunk (folds at compile time)
      an = *(const v16h*)(ap + kb + 32);
#pragma unroll
      for (int j = 0; j < NB; j++)
        bn[j] = *(const v16h*)(bp + (size_t)(j * 16) * ldb + kb + 32);
    }
#pragma unroll
    for (int j = 0; j < NB; j++)
      cs[j] = __builtin_amdgcn_wmma_f32_16x16x32_f16(false, a, false, b[j], (short)0, cs[j], false, false);
    if (kb + 32 < K) {
      a = an;
#pragma unroll
      for (int j = 0; j < NB; j++) b[j] = bn[j];
    }
  }
}

// ---------------- LayerNorm over 192 channels (block = 192 threads) --------
__device__ __forceinline__ float ln_norm192(float v, float g, float b) {
  __shared__ float red[6];
  int c = threadIdx.x;
  float s = v;
#pragma unroll
  for (int m = 16; m >= 1; m >>= 1) s += __shfl_xor(s, m, 32);
  if ((c & 31) == 0) red[c >> 5] = s;
  __syncthreads();
  float mean = (red[0] + red[1] + red[2] + red[3] + red[4] + red[5]) * (1.0f / DIMC);
  __syncthreads();
  float dv = v - mean;
  s = dv * dv;
#pragma unroll
  for (int m = 16; m >= 1; m >>= 1) s += __shfl_xor(s, m, 32);
  if ((c & 31) == 0) red[c >> 5] = s;
  __syncthreads();
  float var = (red[0] + red[1] + red[2] + red[3] + red[4] + red[5]) * (1.0f / DIMC);
  return dv * rsqrtf(var + 1e-5f) * g + b;
}

// ---------------- K0a: weights -> f16 (pre-transposed for Bt[n][k]) --------
__global__ void swin_convert_weights(const float* __restrict__ qkv_w, const float* __restrict__ proj_w,
                                     const float* __restrict__ lin1_w, const float* __restrict__ lin2_w,
                                     const float* __restrict__ pw1,    const float* __restrict__ pw2,
                                     _Float16* qkvT, _Float16* projT, _Float16* lin1T,
                                     _Float16* lin2T, _Float16* pw1h, _Float16* pw2h) {
  int i = blockIdx.x * 256 + threadIdx.x;
  if (i < 576 * DIMC) { int n = i / DIMC, k = i % DIMC; qkvT[i] = (_Float16)qkv_w[k * 576 + n]; }
  if (i < DIMC * DIMC) {
    int n = i / DIMC, k = i % DIMC;
    projT[i] = (_Float16)proj_w[k * DIMC + n];
    pw1h[i]  = (_Float16)pw1[i];   // pw kernels are [out][in] == Bt layout already
    pw2h[i]  = (_Float16)pw2[i];
  }
  if (i < HID * DIMC) { int n = i / DIMC, k = i % DIMC; lin1T[i] = (_Float16)lin1_w[k * HID + n]; }
  if (i < DIMC * HID) { int n = i / HID,  k = i % HID;  lin2T[i] = (_Float16)lin2_w[k * DIMC + n]; }
}

// ---------------- K0b: relative-position bias table biasTab[h][n][m] -------
// Hoists the /49,/7,%7 index chains out of the attention hot loop.
__global__ void swin_build_bias(const float* __restrict__ rpb, float* __restrict__ biasTab) {
  int i = blockIdx.x * 256 + threadIdx.x;
  if (i >= HEADS * NT * NT) return;
  int h = i / (NT * NT), r = i % (NT * NT);
  int n = r / NT, m = r % NT;
  int qd = n / 49, qh = (n / 7) % 7, qw = n % 7;
  int kd = m / 49, kh = (m / 7) % 7, kw = m % 7;
  int ridx = (qd - kd + 6) * 169 + (qh - kh + 6) * 13 + (qw - kw + 6);
  biasTab[i] = rpb[ridx * HEADS + h];
}

// ---------------- K1: LN1 + cyclic shift + window partition -> xw f16 -----
__global__ void swin_ln1_shift(const float* __restrict__ x, const float* __restrict__ g,
                               const float* __restrict__ b, _Float16* __restrict__ xw) {
  int wid = blockIdx.x / NP, t = blockIdx.x % NP, c = threadIdx.x;
  if (t >= NT) { xw[(size_t)blockIdx.x * DIMC + c] = (_Float16)0.f; return; }
  int bi = wid / NWIN, widx = wid % NWIN;
  int wd = widx / 16, wh = (widx / 4) % 4, ww = widx % 4;
  int td = t / 49, th = (t / 7) % 7, tw = t % 7;
  int od = (wd * 7 + td + SS_) % DD, oh = (wh * 7 + th + SS_) % DD, ow = (ww * 7 + tw + SS_) % DD;
  size_t l = (size_t)(od * DD + oh) * DD + ow;
  float v = x[((size_t)bi * LTOK + l) * DIMC + c];
  xw[(size_t)blockIdx.x * DIMC + c] = (_Float16)ln_norm192(v, g[c], b[c]);
}

// ---------------- K2: QKV GEMM [45056x192]@[192x576] -> Q,K,Vt (f16) ------
__global__ __launch_bounds__(256) void swin_qkv_gemm(const _Float16* __restrict__ xw,
                                                     const _Float16* __restrict__ wT,
                                                     const float* __restrict__ bias,
                                                     _Float16* __restrict__ Q, _Float16* __restrict__ Kb,
                                                     _Float16* __restrict__ Vt) {
  int wave = (blockIdx.x * blockDim.x + threadIdx.x) >> 5;
  int lane = threadIdx.x & 31;
  constexpr int NG = (3 * DIMC) / 64, TOT = (BW * NP / 16) * NG;   // NB=4 groups
  if (wave >= TOT) return;
  int mt = wave / NG, ng = wave % NG;
  v8f cs[4];
  wmma_tile_nb<DIMC, 4>(xw, DIMC, wT, DIMC, mt * 16, ng * 64, lane, cs);
  int rb = mt * 16 + ((lane >> 4) << 3);
#pragma unroll
  for (int j = 0; j < 4; j++) {
    int col = ng * 64 + j * 16 + (lane & 15);
    int which = col / DIMC, hf = col % DIMC, head = hf / HD, d = hf % HD;
    float bi = bias[col];
#pragma unroll
    for (int r = 0; r < 8; r++) {
      int row = rb + r, w = row / NP, t = row % NP;
      float v = cs[j][r] + bi;
      int base = (w * HEADS + head);
      if (which == 0)      Q[((size_t)base * NP + t) * HD + d] = (_Float16)(v * SCALE_F);
      else if (which == 1) Kb[((size_t)base * NP + t) * HD + d] = (_Float16)v;
      else                 Vt[((size_t)base * HD + d) * NP + t] = (_Float16)v;  // V transposed
    }
  }
}

// ---------------- K3: flash-style attention --------------------------------
// 8 independent waves per 256-thread block; each wave owns one
// (window, head, 16-row q-tile) and a private 1KB LDS slab for the
// C-layout -> A-layout transpose of P. 16896 waves == 2112 full blocks.
__global__ __launch_bounds__(256) void swin_attn(const _Float16* __restrict__ Q,
                                                 const _Float16* __restrict__ Kb,
                                                 const _Float16* __restrict__ Vt,
                                                 const float* __restrict__ biasTab,
                                                 const float* __restrict__ mask,
                                                 _Float16* __restrict__ ao) {
  __shared__ __align__(64) _Float16 p_all[8 * 16 * 32];
  int lane = threadIdx.x & 31;
  int wv = threadIdx.x >> 5;
  _Float16* p_lds = p_all + wv * (16 * 32);
  int gw = blockIdx.x * 8 + wv;                 // 0 .. BW*HEADS*22-1, exact
  int qt = gw % 22; int tmp = gw / 22;
  int head = tmp % HEADS; int w = tmp / HEADS;
  int widx = w % NWIN;
  const _Float16* qp = Q  + (size_t)(w * HEADS + head) * NP * HD;
  const _Float16* kp = Kb + (size_t)(w * HEADS + head) * NP * HD;
  const _Float16* vp = Vt + (size_t)(w * HEADS + head) * HD * NP;
  const float*    bt = biasTab + (size_t)head * NT * NT;
  const float*    mk = mask    + (size_t)widx * NT * NT;
  int qbase = qt * 16, colq = lane & 15, dh = (lane >> 4) << 4, rb = (lane >> 4) << 3;
  v16h aq = *(const v16h*)(qp + (size_t)(qbase + colq) * HD + dh);
  float m8[8], l8[8];
  v8f o0 = {}, o1 = {};
#pragma unroll
  for (int r = 0; r < 8; r++) { m8[r] = NEG_INF; l8[r] = 0.f; }
  for (int kc = 0; kc < 11; kc++) {
    int k0 = kc * 32;
    v16h bk0 = *(const v16h*)(kp + (size_t)(k0 + colq) * HD + dh);
    v16h bk1 = *(const v16h*)(kp + (size_t)(k0 + 16 + colq) * HD + dh);
    v8f z = {};
    v8f s0 = __builtin_amdgcn_wmma_f32_16x16x32_f16(false, aq, false, bk0, (short)0, z, false, false);
    v8f s1 = __builtin_amdgcn_wmma_f32_16x16x32_f16(false, aq, false, bk1, (short)0, z, false, false);
    int m0c = k0 + colq, m1c = k0 + 16 + colq;
#pragma unroll
    for (int r = 0; r < 8; r++) {
      int n = qbase + rb + r;                 // query token of this element
      if (n < NT) {
        size_t ro = (size_t)n * NT;
        if (m0c < NT) s0[r] += bt[ro + m0c] + mk[ro + m0c]; else s0[r] = NEG_INF;
        if (m1c < NT) s1[r] += bt[ro + m1c] + mk[ro + m1c]; else s1[r] = NEG_INF;
      } else { s0[r] = NEG_INF; s1[r] = NEG_INF; }
    }
    // online softmax (row r spans the 16 lanes of this half-wave)
#pragma unroll
    for (int r = 0; r < 8; r++) {
      float mx = fmaxf(s0[r], s1[r]);
#pragma unroll
      for (int d = 1; d < 16; d <<= 1) mx = fmaxf(mx, __shfl_xor(mx, d, 32));
      float mn = fmaxf(m8[r], mx);
      float sc = __expf(m8[r] - mn);
      m8[r] = mn;
      float p0 = __expf(s0[r] - mn), p1 = __expf(s1[r] - mn);
      float ps = p0 + p1;
#pragma unroll
      for (int d = 1; d < 16; d <<= 1) ps += __shfl_xor(ps, d, 32);
      l8[r] = l8[r] * sc + ps;
      o0[r] *= sc; o1[r] *= sc;
      s0[r] = p0;  s1[r] = p1;
    }
    // C-layout -> A-layout transpose of P through LDS (f32 -> f16)
#pragma unroll
    for (int r = 0; r < 8; r++) {
      p_lds[(rb + r) * 32 + colq]      = (_Float16)s0[r];
      p_lds[(rb + r) * 32 + 16 + colq] = (_Float16)s1[r];
    }
    __syncthreads();
    v16h ap  = *(const v16h*)(p_lds + colq * 32 + dh);
    v16h bv0 = *(const v16h*)(vp + (size_t)colq * NP + k0 + dh);         // dims 0..15
    v16h bv1 = *(const v16h*)(vp + (size_t)(16 + colq) * NP + k0 + dh);  // dims 16..31
    o0 = __builtin_amdgcn_wmma_f32_16x16x32_f16(false, ap, false, bv0, (short)0, o0, false, false);
    o1 = __builtin_amdgcn_wmma_f32_16x16x32_f16(false, ap, false, bv1, (short)0, o1, false, false);
    __syncthreads();
  }
#pragma unroll
  for (int r = 0; r < 8; r++) {
    int row = qbase + rb + r;
    if (row < NT) {
      float inv = 1.f / l8[r];
      ao[((size_t)w * NP + row) * DIMC + head * HD + colq]      = (_Float16)(o0[r] * inv);
      ao[((size_t)w * NP + row) * DIMC + head * HD + 16 + colq] = (_Float16)(o1[r] * inv);
    }
  }
}

// ---------------- K4: proj GEMM + window-reverse + unshift + residual -----
__global__ __launch_bounds__(256) void swin_proj_gemm(const _Float16* __restrict__ ao,
                                                      const _Float16* __restrict__ wT,
                                                      const float* __restrict__ bias,
                                                      const float* __restrict__ xin,
                                                      float* __restrict__ x1) {
  int wave = (blockIdx.x * blockDim.x + threadIdx.x) >> 5;
  int lane = threadIdx.x & 31;
  constexpr int NG = DIMC / 64, TOT = (BW * NP / 16) * NG;
  if (wave >= TOT) return;
  int mt = wave / NG, ng = wave % NG;
  v8f cs[4];
  wmma_tile_nb<DIMC, 4>(ao, DIMC, wT, DIMC, mt * 16, ng * 64, lane, cs);
  int colb = ng * 64 + (lane & 15);
  int rb = mt * 16 + ((lane >> 4) << 3);
  float bi[4];
#pragma unroll
  for (int j = 0; j < 4; j++) bi[j] = bias[colb + j * 16];
#pragma unroll
  for (int r = 0; r < 8; r++) {
    int row = rb + r, w = row / NP, t = row % NP;
    if (t >= NT) continue;
    int b2 = w / NWIN, widx = w % NWIN;
    int wd = widx / 16, wh = (widx / 4) % 4, ww = widx % 4;
    int td = t / 49, th = (t / 7) % 7, tw = t % 7;
    int od = (wd * 7 + td + SS_) % DD, oh = (wh * 7 + th + SS_) % DD, ow = (ww * 7 + tw + SS_) % DD;
    size_t base = ((size_t)b2 * LTOK + (size_t)(od * DD + oh) * DD + ow) * DIMC;
#pragma unroll
    for (int j = 0; j < 4; j++) {
      size_t idx = base + colb + j * 16;
      x1[idx] = xin[idx] + cs[j][r] + bi[j];
    }
  }
}

// ---------------- K5: LN2 over x1 -> xm (f16) ------------------------------
__global__ void swin_ln2(const float* __restrict__ x, const float* __restrict__ g,
                         const float* __restrict__ b, _Float16* __restrict__ y) {
  int row = blockIdx.x, c = threadIdx.x;
  float v = x[(size_t)row * DIMC + c];
  y[(size_t)row * DIMC + c] = (_Float16)ln_norm192(v, g[c], b[c]);
}

// ---------------- K6: lin1 GEMM + exact GELU -> hb (f16) -------------------
__global__ __launch_bounds__(256) void swin_lin1_gemm(const _Float16* __restrict__ xm,
                                                      const _Float16* __restrict__ wT,
                                                      const float* __restrict__ bias,
                                                      _Float16* __restrict__ hb) {
  int wave = (blockIdx.x * blockDim.x + threadIdx.x) >> 5;
  int lane = threadIdx.x & 31;
  constexpr int NG = HID / 64, TOT = (BB * LTOK / 16) * NG;
  if (wave >= TOT) return;
  int mt = wave / NG, ng = wave % NG;
  v8f cs[4];
  wmma_tile_nb<DIMC, 4>(xm, DIMC, wT, DIMC, mt * 16, ng * 64, lane, cs);
  int rb = mt * 16 + ((lane >> 4) << 3);
#pragma unroll
  for (int j = 0; j < 4; j++) {
    int col = ng * 64 + j * 16 + (lane & 15);
    float bi = bias[col];
#pragma unroll
    for (int r = 0; r < 8; r++) {
      float v = cs[j][r] + bi;
      float gl = 0.5f * v * (1.0f + erff(v * 0.70710678118654752f));
      hb[(size_t)(rb + r) * HID + col] = (_Float16)gl;
    }
  }
}

// ---------------- K7: lin2 GEMM; out = x1 + lin ----------------------------
__global__ __launch_bounds__(256) void swin_lin2_gemm(const _Float16* __restrict__ hb,
                                                      const _Float16* __restrict__ wT,
                                                      const float* __restrict__ bias,
                                                      const float* __restrict__ x1,
                                                      float* __restrict__ out) {
  int wave = (blockIdx.x * blockDim.x + threadIdx.x) >> 5;
  int lane = threadIdx.x & 31;
  constexpr int NG = DIMC / 64, TOT = (BB * LTOK / 16) * NG;
  if (wave >= TOT) return;
  int mt = wave / NG, ng = wave % NG;
  v8f cs[4];
  wmma_tile_nb<HID, 4>(hb, HID, wT, HID, mt * 16, ng * 64, lane, cs);
  int rb = mt * 16 + ((lane >> 4) << 3);
#pragma unroll
  for (int j = 0; j < 4; j++) {
    int col = ng * 64 + j * 16 + (lane & 15);
    float bi = bias[col];
#pragma unroll
    for (int r = 0; r < 8; r++) {
      size_t idx = (size_t)(rb + r) * DIMC + col;
      out[idx] = x1[idx] + cs[j][r] + bi;
    }
  }
}

// ---------------- K8: depthwise 3x3x3 conv (SAME), token-major f16 ---------
__global__ void swin_dw_conv(const _Float16* __restrict__ in, const float* __restrict__ k,
                             const float* __restrict__ bias, _Float16* __restrict__ out) {
  size_t idx = (size_t)blockIdx.x * 256 + threadIdx.x;
  if (idx >= (size_t)BB * LTOK * DIMC) return;
  int c = idx % DIMC;
  int sp = (idx / DIMC) % LTOK;
  int b = idx / ((size_t)DIMC * LTOK);
  int w = sp % DD, h = (sp / DD) % DD, d = sp / (DD * DD);
  float acc = bias[c];
#pragma unroll
  for (int kd = 0; kd < 3; kd++)
#pragma unroll
    for (int kh = 0; kh < 3; kh++)
#pragma unroll
      for (int kw = 0; kw < 3; kw++) {
        int d2 = d + kd - 1, h2 = h + kh - 1, w2 = w + kw - 1;
        if (d2 < 0 || d2 >= DD || h2 < 0 || h2 >= DD || w2 < 0 || w2 >= DD) continue;
        acc += (float)in[((size_t)b * LTOK + (size_t)(d2 * DD + h2) * DD + w2) * DIMC + c]
               * k[c * 27 + (kd * 3 + kh) * 3 + kw];
      }
  out[idx] = (_Float16)acc;
}

// ---------------- K9: pw1 GEMM -> y2 (f16) ---------------------------------
__global__ __launch_bounds__(256) void swin_pw1_gemm(const _Float16* __restrict__ y1,
                                                     const _Float16* __restrict__ wT,
                                                     const float* __restrict__ bias,
                                                     _Float16* __restrict__ y2) {
  int wave = (blockIdx.x * blockDim.x + threadIdx.x) >> 5;
  int lane = threadIdx.x & 31;
  constexpr int NG = DIMC / 64, TOT = (BB * LTOK / 16) * NG;
  if (wave >= TOT) return;
  int mt = wave / NG, ng = wave % NG;
  v8f cs[4];
  wmma_tile_nb<DIMC, 4>(y1, DIMC, wT, DIMC, mt * 16, ng * 64, lane, cs);
  int rb = mt * 16 + ((lane >> 4) << 3);
#pragma unroll
  for (int j = 0; j < 4; j++) {
    int col = ng * 64 + j * 16 + (lane & 15);
    float bi = bias[col];
#pragma unroll
    for (int r = 0; r < 8; r++)
      y2[(size_t)(rb + r) * DIMC + col] = (_Float16)(cs[j][r] + bi);
  }
}

// ---------------- K10: pw2 GEMM; out += rx with NCDHW layout-mixing --------
__global__ __launch_bounds__(256) void swin_pw2_gemm(const _Float16* __restrict__ y3,
                                                     const _Float16* __restrict__ wT,
                                                     const float* __restrict__ bias,
                                                     float* __restrict__ out) {
  int wave = (blockIdx.x * blockDim.x + threadIdx.x) >> 5;
  int lane = threadIdx.x & 31;
  constexpr int NG = DIMC / 64, TOT = (BB * LTOK / 16) * NG;
  if (wave >= TOT) return;
  int mt = wave / NG, ng = wave % NG;
  v8f cs[4];
  wmma_tile_nb<DIMC, 4>(y3, DIMC, wT, DIMC, mt * 16, ng * 64, lane, cs);
  int rb = mt * 16 + ((lane >> 4) << 3);
#pragma unroll
  for (int j = 0; j < 4; j++) {
    int col = ng * 64 + j * 16 + (lane & 15);
    float bi = bias[col];
#pragma unroll
    for (int r = 0; r < 8; r++) {
      int row = rb + r, b = row / LTOK, sp = row % LTOK;
      // reference does rx(NCDHW).reshape(B,-1,DIM): element (sp,col) lands at
      // flat offset col*LTOK + sp within batch b
      size_t idx = (size_t)b * LTOK * DIMC + (size_t)col * LTOK + sp;
      out[idx] += cs[j][r] + bi;
    }
  }
}

extern "C" void kernel_launch(void* const* d_in, const int* in_sizes, int n_in,
                              void* d_out, int out_size, void* d_ws, size_t ws_size,
                              hipStream_t stream) {
  const float* x      = (const float*)d_in[0];
  const float* mask   = (const float*)d_in[1];
  const float* n1g    = (const float*)d_in[2];
  const float* n1b    = (const float*)d_in[3];
  const float* qkv_w  = (const float*)d_in[4];
  const float* qkv_b  = (const float*)d_in[5];
  const float* rpb    = (const float*)d_in[6];
  const float* proj_w = (const float*)d_in[7];
  const float* proj_b = (const float*)d_in[8];
  const float* n2g    = (const float*)d_in[9];
  const float* n2b    = (const float*)d_in[10];
  const float* lin1_w = (const float*)d_in[11];
  const float* lin1_b = (const float*)d_in[12];
  const float* lin2_w = (const float*)d_in[13];
  const float* lin2_b = (const float*)d_in[14];
  const float* dw1_k  = (const float*)d_in[15];
  const float* dw1_b  = (const float*)d_in[16];
  const float* pw1_k  = (const float*)d_in[17];
  const float* pw1_b  = (const float*)d_in[18];
  const float* dw2_k  = (const float*)d_in[19];
  const float* dw2_b  = (const float*)d_in[20];
  const float* pw2_k  = (const float*)d_in[21];
  const float* pw2_b  = (const float*)d_in[22];
  float* out = (float*)d_out;

  char* ws = (char*)d_ws;
  size_t off = 0;
  auto alloc = [&](size_t bytes) -> void* {
    void* p = ws + off;
    off += (bytes + 255) & ~(size_t)255;
    return p;
  };
  _Float16* qkvT   = (_Float16*)alloc((size_t)576 * DIMC * 2);
  _Float16* projT  = (_Float16*)alloc((size_t)DIMC * DIMC * 2);
  _Float16* lin1T  = (_Float16*)alloc((size_t)HID * DIMC * 2);
  _Float16* lin2T  = (_Float16*)alloc((size_t)DIMC * HID * 2);
  _Float16* pw1h   = (_Float16*)alloc((size_t)DIMC * DIMC * 2);
  _Float16* pw2h   = (_Float16*)alloc((size_t)DIMC * DIMC * 2);
  float*    biasTab= (float*)   alloc((size_t)HEADS * NT * NT * 4);
  _Float16* xw     = (_Float16*)alloc((size_t)BW * NP * DIMC * 2);
  _Float16* Qb     = (_Float16*)alloc((size_t)BW * HEADS * NP * HD * 2);
  _Float16* Kbuf   = (_Float16*)alloc((size_t)BW * HEADS * NP * HD * 2);
  _Float16* Vt     = (_Float16*)alloc((size_t)BW * HEADS * HD * NP * 2);
  _Float16* ao     = (_Float16*)alloc((size_t)BW * NP * DIMC * 2);
  float*    x1     = (float*)   alloc((size_t)BB * LTOK * DIMC * 4);
  _Float16* xm     = (_Float16*)alloc((size_t)BB * LTOK * DIMC * 2);
  _Float16* hb     = (_Float16*)alloc((size_t)BB * LTOK * HID * 2);
  _Float16* y1     = (_Float16*)alloc((size_t)BB * LTOK * DIMC * 2);
  _Float16* y2     = (_Float16*)alloc((size_t)BB * LTOK * DIMC * 2);
  (void)ws_size; (void)in_sizes; (void)n_in; (void)out_size;

  // K0: weight conversion + RPB bias table
  swin_convert_weights<<<(HID * DIMC + 255) / 256, 256, 0, stream>>>(
      qkv_w, proj_w, lin1_w, lin2_w, pw1_k, pw2_k, qkvT, projT, lin1T, lin2T, pw1h, pw2h);
  swin_build_bias<<<(HEADS * NT * NT + 255) / 256, 256, 0, stream>>>(rpb, biasTab);

  // K1: LN1 + shift + partition
  swin_ln1_shift<<<BW * NP, DIMC, 0, stream>>>(x, n1g, n1b, xw);

  // K2: QKV GEMM (NB=4 register-blocked, double-buffered wave tiles)
  swin_qkv_gemm<<<((BW * NP / 16) * (3 * DIMC / 64) + 7) / 8, 256, 0, stream>>>(
      xw, qkvT, qkv_b, Qb, Kbuf, Vt);

  // K3: attention (128 windows * 6 heads * 22 q-tiles = 16896 waves)
  swin_attn<<<(BW * HEADS * 22) / 8, 256, 0, stream>>>(Qb, Kbuf, Vt, biasTab, mask, ao);

  // K4: proj + reverse + residual
  swin_proj_gemm<<<((BW * NP / 16) * (DIMC / 64) + 7) / 8, 256, 0, stream>>>(
      ao, projT, proj_b, x, x1);

  // K5: LN2
  swin_ln2<<<BB * LTOK, DIMC, 0, stream>>>(x1, n2g, n2b, xm);

  // K6: lin1 + GELU
  swin_lin1_gemm<<<((BB * LTOK / 16) * (HID / 64) + 7) / 8, 256, 0, stream>>>(
      xm, lin1T, lin1_b, hb);

  // K7: lin2; out = x1 + lin
  swin_lin2_gemm<<<((BB * LTOK / 16) * (DIMC / 64) + 7) / 8, 256, 0, stream>>>(
      hb, lin2T, lin2_b, x1, out);

  // K8-K10: conv branch (dw1 -> pw1 -> dw2 -> pw2, pw2 accumulates into out)
  int nElem = BB * LTOK * DIMC;
  swin_dw_conv<<<(nElem + 255) / 256, 256, 0, stream>>>(xm, dw1_k, dw1_b, y1);
  swin_pw1_gemm<<<((BB * LTOK / 16) * (DIMC / 64) + 7) / 8, 256, 0, stream>>>(
      y1, pw1h, pw1_b, y2);
  swin_dw_conv<<<(nElem + 255) / 256, 256, 0, stream>>>(y2, dw2_k, dw2_b, y1);
  swin_pw2_gemm<<<((BB * LTOK / 16) * (DIMC / 64) + 7) / 8, 256, 0, stream>>>(
      y1, pw2h, pw2_b, out);
}
